// SNN_FiLM_9732395893348
// MI455X (gfx1250) — matrix operand order repaired and checked by reading, not compiled, not run on hardware
//
#include <hip/hip_runtime.h>
#include <hip/hip_bf16.h>

// CDNA5 / gfx1250 vector types (true clang vectors, so nontemporal builtins work)
typedef __attribute__((ext_vector_type(16))) _Float16 v16h;
typedef __attribute__((ext_vector_type(8)))  _Float16 v8h;
typedef __attribute__((ext_vector_type(8)))  float    v8f;
typedef __attribute__((ext_vector_type(4)))  float    f4;

// Problem constants (from setup_inputs)
#define TT 16
#define BB 16
#define CC 256     // channels C
#define HH 32
#define WW 32
#define KCC 256    // Cc (LIF / GEMM K dim)
#define MROWS (TT*BB)     // 256 GEMM rows
#define NCOLS (2*CC)      // 512 GEMM cols
#define HW (HH*WW)        // 1024 elems per (t,b,c)

// ---------------------------------------------------------------------------
// Kernel A: split W (f32 [512][256]) into hi/lo f16 halves for exact 2-term WMMA
// ---------------------------------------------------------------------------
__global__ void split_w_kernel(const float* __restrict__ W,
                               _Float16* __restrict__ whi,
                               _Float16* __restrict__ wlo) {
    int i = blockIdx.x * blockDim.x + threadIdx.x;   // 0 .. 512*256-1
    float w = W[i];
    _Float16 hi = (_Float16)w;
    whi[i] = hi;
    wlo[i] = (_Float16)(w - (float)hi);
}

// ---------------------------------------------------------------------------
// Kernel B: hard-reset LIF scan over T; spikes written as f16 in [m=t*B+b][cc]
// v' = (v + x)/2 ; s = (v' >= 1) ; v = s ? 0 : v'
// ---------------------------------------------------------------------------
__global__ void lif_kernel(const float* __restrict__ cond,
                           _Float16* __restrict__ spk) {
    int tid = blockIdx.x * blockDim.x + threadIdx.x; // 0 .. B*Cc-1 (4096)
    float v = 0.0f;
    #pragma unroll
    for (int t = 0; t < TT; ++t) {
        int idx = t * (BB * KCC) + tid;
        v = 0.5f * (v + cond[idx]);
        float s = (v >= 1.0f) ? 1.0f : 0.0f;
        v = (s != 0.0f) ? 0.0f : v;
        spk[idx] = (_Float16)s;                      // exact 0/1 in f16
    }
}

// ---------------------------------------------------------------------------
// Kernel C: params[256][512] = spikes @ W^T + b via V_WMMA_F32_16X16X32_F16.
// One wave per 16x16 output tile. Two WMMAs per K-block (lo+hi W split)
// recover near-f32 precision since A is exactly representable.
// ---------------------------------------------------------------------------
__global__ void gemm_wmma_kernel(const _Float16* __restrict__ spk,
                                 const _Float16* __restrict__ whi,
                                 const _Float16* __restrict__ wlo,
                                 const float*    __restrict__ bias,
                                 float*          __restrict__ params) {
    const int wave  = blockIdx.x * (blockDim.x >> 5) + (threadIdx.x >> 5);
    const int lane  = threadIdx.x & 31;
    const int mtile = wave >> 5;          // 0..15  (M tiles)
    const int ntile = wave & 31;          // 0..31  (N tiles)
    const int row   = lane & 15;          // A row within tile; also D column
    const int hi    = lane >> 4;          // lane half select
    const int dcol  = ntile * 16 + row;   // global output column (0..511)

    const _Float16* __restrict__ arow  = spk + (mtile * 16 + row) * KCC;
    const _Float16* __restrict__ bhrow = whi + dcol * KCC;
    const _Float16* __restrict__ blrow = wlo + dcol * KCC;

    v8f acc = {};
    #pragma unroll
    for (int kb = 0; kb < KCC / 32; ++kb) {
        // A fragment: a[j] = spk[row][kb*32 + hi*16 + j] (matches B k order)
        const int k0 = kb * 32 + hi * 16;
        v8h a0 = *(const v8h*)(arow + k0);        // 16B aligned
        v8h a1 = *(const v8h*)(arow + k0 + 8);    // 16B aligned
        v16h a;
        #pragma unroll
        for (int j = 0; j < 8; ++j) { a[j] = a0[j]; a[8 + j] = a1[j]; }

        // B fragment (contiguous k, same order as A): b[j] = W[dcol][kb*32 + hi*16 + j]
        const int kb2 = kb * 32 + hi * 16;
        v16h bh = *(const v16h*)(bhrow + kb2);    // 32B aligned
        v16h bl = *(const v16h*)(blrow + kb2);    // 32B aligned

        acc = __builtin_amdgcn_wmma_f32_16x16x32_f16(false, a, false, bl,
                                                     (short)0, acc, false, false);
        acc = __builtin_amdgcn_wmma_f32_16x16x32_f16(false, a, false, bh,
                                                     (short)0, acc, false, false);
    }

    const float bv = bias[dcol];
    #pragma unroll
    for (int r = 0; r < 8; ++r) {
        // D layout: acc[r] -> M = r + 8*hi, N = row
        const int m = mtile * 16 + r + hi * 8;
        params[m * NCOLS + dcol] = acc[r] + bv;
    }
}

// ---------------------------------------------------------------------------
// Kernel D: FiLM apply, bandwidth bound. out = (gamma+1)*x + beta.
// One block per (t,b,c): 256 threads x float4 covers H*W=1024 floats.
// Non-temporal b128 load/store: 512MB stream, don't pollute L2.
// ---------------------------------------------------------------------------
__global__ void film_kernel(const f4* __restrict__ x,
                            const float* __restrict__ params,
                            f4* __restrict__ out) {
    const int tbc = blockIdx.x;          // (t*B + b)*C + c, 0..65535
    const int tb  = tbc >> 8;            // C = 256
    const int c   = tbc & 255;

    // block-uniform -> scalar loads
    const float g  = params[tb * NCOLS + c] + 1.0f;
    const float be = params[tb * NCOLS + CC + c];

    const int i = tbc * (HW / 4) + threadIdx.x;     // float4 index
    f4 v = __builtin_nontemporal_load(&x[i]);
    f4 o;
    o.x = g * v.x + be;
    o.y = g * v.y + be;
    o.z = g * v.z + be;
    o.w = g * v.w + be;
    __builtin_nontemporal_store(o, &out[i]);
}

// ---------------------------------------------------------------------------
extern "C" void kernel_launch(void* const* d_in, const int* in_sizes, int n_in,
                              void* d_out, int out_size, void* d_ws, size_t ws_size,
                              hipStream_t stream) {
    const float* x    = (const float*)d_in[0];   // [T,B,C,H,W]
    const float* cond = (const float*)d_in[1];   // [T,B,Cc]
    const float* W    = (const float*)d_in[2];   // [2C,Cc]
    const float* bias = (const float*)d_in[3];   // [2C]
    float* out        = (float*)d_out;

    // Workspace carve-up (all offsets 256B-aligned)
    char* ws = (char*)d_ws;
    _Float16* spk   = (_Float16*)(ws);                       // 256*256 f16 = 128KB
    _Float16* whi   = (_Float16*)(ws + (128 << 10));         // 512*256 f16 = 256KB
    _Float16* wlo   = (_Float16*)(ws + (384 << 10));         // 512*256 f16 = 256KB
    float*    prm   = (float*)   (ws + (640 << 10));         // 256*512 f32 = 512KB

    // A: W split (512*256 = 131072 elements)
    split_w_kernel<<<NCOLS * KCC / 256, 256, 0, stream>>>(W, whi, wlo);

    // B: LIF scan (B*Cc = 4096 threads)
    lif_kernel<<<BB * KCC / 256, 256, 0, stream>>>(cond, spk);

    // C: WMMA GEMM: 16*32 = 512 waves, 4 waves/block
    gemm_wmma_kernel<<<128, 128, 0, stream>>>(spk, whi, wlo, bias, prm);

    // D: FiLM apply: one block per (t,b,c)
    film_kernel<<<TT * BB * CC, 256, 0, stream>>>((const f4*)x, prm, (f4*)out);
}
